// MIPloss_80341658239322
// MI455X (gfx1250) — compile-verified, matrix operand order checked by reading, not run on hardware
//
#include <hip/hip_runtime.h>

// MI455X (gfx1250) MIP-loss. Memory-bound (~0.5 GB traffic @ 23.3 TB/s).
// SSIM Gaussian blurs run on the WMMA units (v_wmma_f32_16x16x32_f16).
// All LDS operand traffic is laid out fragment-contiguous so every WMMA
// operand is 2x ds_load_b128 and every C-tile spill is packed b128 stores.

typedef __attribute__((ext_vector_type(16))) _Float16 v16h;
typedef __attribute__((ext_vector_type(8)))  _Float16 v8h;
typedef __attribute__((ext_vector_type(8)))  float    v8f;

#define S_SLICES 64
#define IMG_HW   65536      // 256*256
#define IMG_W    256
#define N_IMG    128        // B*S = 2*64
#define N_ELEM   8388608u   // 2*64*256*256
#define OUT_HW   246
#define GEPS     1e-20f

// normalized gaussian(11, sigma=1.5): g[d] = exp(-(d-5)^2/4.5)/sum
__device__ __forceinline__ float gauss_w(int d) {
  float x = (float)(d - 5);
  return 0.26601725f * __expf(-x * x * (1.0f / 4.5f));
}

__device__ __forceinline__ v16h frag_cat(v8h lo, v8h hi) {
  return __builtin_shufflevector(lo, hi, 0, 1, 2, 3, 4, 5, 6, 7,
                                 8, 9, 10, 11, 12, 13, 14, 15);
}

// ---------------------------------------------------------------------------
// Kernel 1: zero the 6 accumulators (3 L1 sums, 3 SSIM-map sums)
// ---------------------------------------------------------------------------
__global__ void zero_kernel(float* acc) {
  if (threadIdx.x < 6) acc[threadIdx.x] = 0.0f;
}

// ---------------------------------------------------------------------------
// Kernel 2: per-pixel scan over the 64 slices (prefix/suffix gumbel-softmax
// MIP + cummax), plus all three L1 sums. x and x+gumbel live in registers.
// ---------------------------------------------------------------------------
__global__ __launch_bounds__(256)
void scan_kernel(const float* __restrict__ xf, const float* __restrict__ tg,
                 const float* __restrict__ un,
                 float* __restrict__ p1, float* __restrict__ t1,
                 float* __restrict__ p2, float* __restrict__ t2,
                 float* __restrict__ acc) {
  unsigned pix = blockIdx.x * 256u + threadIdx.x;             // 0..131071
  size_t base = (size_t)(pix >> 16) * ((size_t)S_SLICES * IMG_HW) + (pix & 65535u);

  float xv[S_SLICES], yv[S_SLICES];
  float m = -1e30f;
#pragma unroll
  for (int s = 0; s < S_SLICES; ++s) {
    size_t off = base + (size_t)s * IMG_HW;
    float x = xf[off];
    float u = un[off];
    float g = -__logf(-__logf(u + GEPS) + GEPS);
    xv[s] = x;
    float y = x + g;
    yv[s] = y;
    m = fmaxf(m, y);
  }

  __builtin_prefetch(&tg[base], 0, 0);          // global_prefetch_b8
  float num = 0.f, den = 0.f, tmax = -1e30f;
  float l0 = 0.f, l1s = 0.f, l2s = 0.f;
#pragma unroll
  for (int s = 0; s < S_SLICES; ++s) {
    size_t off = base + (size_t)s * IMG_HW;
    float e = __expf((yv[s] - m) * 2.0f);                     // 1/TEMP = 2
    num += e * xv[s];
    den += e;
    float p = num / den;
    float t = tg[off];
    tmax = fmaxf(tmax, t);
    p1[off] = p;
    t1[off] = tmax;
    l0  += fabsf(xv[s] - t);
    l1s += fabsf(p - tmax);
  }

  num = 0.f; den = 0.f; tmax = -1e30f;
#pragma unroll
  for (int k = 0; k < S_SLICES; ++k) {
    int s = S_SLICES - 1 - k;
    size_t offs = base + (size_t)s * IMG_HW;
    size_t offk = base + (size_t)k * IMG_HW;
    float e = __expf((yv[s] - m) * 2.0f);
    num += e * xv[s];
    den += e;
    float p = num / den;
    float t = tg[offs];
    tmax = fmaxf(tmax, t);
    p2[offk] = p;
    t2[offk] = tmax;
    l2s += fabsf(p - tmax);
  }

  __shared__ float r0[256], r1[256], r2[256];
  r0[threadIdx.x] = l0; r1[threadIdx.x] = l1s; r2[threadIdx.x] = l2s;
  __syncthreads();
  for (int st = 128; st > 0; st >>= 1) {
    if ((int)threadIdx.x < st) {
      r0[threadIdx.x] += r0[threadIdx.x + st];
      r1[threadIdx.x] += r1[threadIdx.x + st];
      r2[threadIdx.x] += r2[threadIdx.x + st];
    }
    __syncthreads();
  }
  if (threadIdx.x == 0) {
    atomicAdd(&acc[0], r0[0]);
    atomicAdd(&acc[1], r1[0]);
    atomicAdd(&acc[2], r2[0]);
  }
}

// ---------------------------------------------------------------------------
// Kernel 3: fused SSIM. One block = one image x one 16-row output strip.
// 5 fields blurred via WMMA (H-pass then V-pass). All operand layouts are
// fragment-contiguous: A/B fragments = 2x ds_load_b128, C spills = b128.
// ---------------------------------------------------------------------------
__global__ __launch_bounds__(256)
void ssim_kernel(const float* __restrict__ xf, const float* __restrict__ tg,
                 const float* __restrict__ p1, const float* __restrict__ t1,
                 const float* __restrict__ p2, const float* __restrict__ t2,
                 float* __restrict__ acc) {
  __shared__ __attribute__((aligned(16))) _Float16 sIn[32][256];      // field tile
  __shared__ __attribute__((aligned(16))) _Float16 sHt[256][32];      // H-blur, [col][row]
  __shared__ __attribute__((aligned(16))) _Float16 sBhFrag[3][32][16];// weights, frag order
  __shared__ __attribute__((aligned(16))) float    sBlurT[5][256][16];// blurred, [f][col][row]
  __shared__ float sRed[256];

  int tid = threadIdx.x;
  unsigned bid = blockIdx.x;
  int p = (int)(bid / (N_IMG * 16));
  unsigned rem = bid % (N_IMG * 16);
  int img = (int)(rem >> 4);
  int strip = (int)(rem & 15);
  int r0 = strip * 16;

  const float* Aimg = (p == 0) ? xf : (p == 1) ? p1 : p2;
  const float* Bimg = (p == 0) ? tg : (p == 1) ? t1 : t2;
  size_t ibase = (size_t)img * IMG_HW;

  // Banded gaussian in per-lane fragment order:
  // logical B[kk][n] = g[kk - n + (sel-1)*16]; lane holds n = lane&15,
  // kk = ((e<8)?e:e+8) + ((lane>>4)<<3).
  for (int i = tid; i < 3 * 32 * 16; i += 256) {
    int sel  = i >> 9;
    int lane = (i >> 4) & 31;
    int e    = i & 15;
    int n  = lane & 15;
    int kk = ((e < 8) ? e : (e + 8)) + ((lane >> 4) << 3);
    int d  = kk - n + (sel - 1) * 16;
    sBhFrag[sel][lane][e] = (_Float16)((d >= 0 && d <= 10) ? gauss_w(d) : 0.0f);
  }
  __syncthreads();

  int lane = tid & 31;
  int wv = tid >> 5;
  int ln16 = lane & 15;
  int kadd = (lane >> 4) << 3;

  for (int f = 0; f < 5; ++f) {
    // ---- load 32x256 field tile (rows clamped; garbage rows masked later)
    for (int i = tid; i < 32 * 256; i += 256) {
      int r = i >> 8, c = i & 255;
      int gr = r0 + r; if (gr > 255) gr = 255;
      size_t off = ibase + (size_t)gr * IMG_W + c;
      float v;
      if      (f == 0) v = Aimg[off];
      else if (f == 1) v = Bimg[off];
      else if (f == 2) { float a = Aimg[off]; v = a * a; }
      else if (f == 3) { float b = Bimg[off]; v = b * b; }
      else             v = Aimg[off] * Bimg[off];
      sIn[r][c] = (_Float16)v;
    }
    __syncthreads();

    // ---- horizontal blur: C(16x16) = rows(16xK) * band(Kx16), K band-pruned
    for (int job = wv; job < 32; job += 8) {     // 2 M-tiles x 16 N-tiles
      int mt = job >> 4;
      int j  = job & 15;
      int row = (mt << 4) + ln16;
      v8f cacc = {0.f, 0.f, 0.f, 0.f, 0.f, 0.f, 0.f, 0.f};
      int nch, kc0, sel0;
      if ((j & 1) == 0)      { nch = 1; kc0 = 16 * j;      sel0 = 1; }
      else if (j == 15)      { nch = 1; kc0 = 16 * j - 16; sel0 = 0; }
      else                   { nch = 2; kc0 = 16 * j - 16; sel0 = 0; }
      for (int q = 0; q < nch; ++q) {
        int kc = kc0 + 32 * q;
        int sel = sel0 + 2 * q;
        v16h av = frag_cat(*(const v8h*)&sIn[row][kc + kadd],
                           *(const v8h*)&sIn[row][kc + kadd + 16]);
        v16h bv = frag_cat(*(const v8h*)&sBhFrag[sel][lane][0],
                           *(const v8h*)&sBhFrag[sel][lane][8]);
        cacc = __builtin_amdgcn_wmma_f32_16x16x32_f16(
            false, av, false, bv, (short)0, cacc, false, false);
      }
      // transposed store: one packed 16B ds_store per lane
      int N = 16 * j + ln16;
      int mb = (mt << 4) + kadd;
      v8h hv;
#pragma unroll
      for (int e = 0; e < 8; ++e) hv[e] = (_Float16)cacc[e];
      *(v8h*)&sHt[N][mb] = hv;
    }
    __syncthreads();

    // ---- vertical blur: C(16x16) = band(16x32) * Hrows(32x16)
    for (int j = wv; j < 16; j += 8) {
      int N = 16 * j + ln16;
      v16h av = frag_cat(*(const v8h*)&sBhFrag[1][lane][0],
                         *(const v8h*)&sBhFrag[1][lane][8]);
      v16h bv = frag_cat(*(const v8h*)&sHt[N][kadd],
                         *(const v8h*)&sHt[N][kadd + 16]);
      v8f cacc = {0.f, 0.f, 0.f, 0.f, 0.f, 0.f, 0.f, 0.f};
      cacc = __builtin_amdgcn_wmma_f32_16x16x32_f16(
          false, av, false, bv, (short)0, cacc, false, false);
      *(v8f*)&sBlurT[f][N][kadd] = cacc;          // contiguous 32B store
    }
    __syncthreads();
  }

  // ---- SSIM map over the valid 246x246 region of this strip, then reduce
  float part = 0.f;
  for (int i = tid; i < 16 * 256; i += 256) {
    int r = i >> 8, c = i & 255;
    if (c < OUT_HW && (r0 + r) < OUT_HW) {
      float mu1 = sBlurT[0][c][r];
      float mu2 = sBlurT[1][c][r];
      float mu1s = mu1 * mu1, mu2s = mu2 * mu2, m12 = mu1 * mu2;
      float s1  = sBlurT[2][c][r] - mu1s;
      float s2  = sBlurT[3][c][r] - mu2s;
      float s12 = sBlurT[4][c][r] - m12;
      const float C1 = 1e-4f, C2 = 9e-4f;
      part += ((2.f * m12 + C1) * (2.f * s12 + C2)) /
              ((mu1s + mu2s + C1) * (s1 + s2 + C2));
    }
  }
  sRed[tid] = part;
  __syncthreads();
  for (int st = 128; st > 0; st >>= 1) {
    if (tid < st) sRed[tid] += sRed[tid + st];
    __syncthreads();
  }
  if (tid == 0) atomicAdd(&acc[3 + p], sRed[0]);
}

// ---------------------------------------------------------------------------
// Kernel 4: combine to scalar loss
// ---------------------------------------------------------------------------
__global__ void final_kernel(const float* __restrict__ acc, float* __restrict__ out) {
  if (threadIdx.x == 0 && blockIdx.x == 0) {
    const float invL1   = 1.0f / 8388608.0f;       // 2*64*256*256
    const float invSSIM = 1.0f / 7746048.0f;       // 128*246*246
    float loss = 0.f;
#pragma unroll
    for (int p = 0; p < 3; ++p)
      loss += acc[p] * invL1 + 1.0f - acc[3 + p] * invSSIM;
    out[0] = loss;                                  // LAMDA == 1
  }
}

// ---------------------------------------------------------------------------
extern "C" void kernel_launch(void* const* d_in, const int* in_sizes, int n_in,
                              void* d_out, int out_size, void* d_ws, size_t ws_size,
                              hipStream_t stream) {
  (void)in_sizes; (void)n_in; (void)out_size; (void)ws_size;
  const float* img_fake = (const float*)d_in[0];
  const float* target   = (const float*)d_in[1];
  const float* u_noise  = (const float*)d_in[2];

  float* p1  = (float*)d_ws;
  float* t1  = p1 + N_ELEM;
  float* p2  = t1 + N_ELEM;
  float* t2  = p2 + N_ELEM;
  float* acc = t2 + N_ELEM;                         // 6 floats

  zero_kernel<<<1, 64, 0, stream>>>(acc);
  scan_kernel<<<131072 / 256, 256, 0, stream>>>(img_fake, target, u_noise,
                                                p1, t1, p2, t2, acc);
  ssim_kernel<<<3 * N_IMG * 16, 256, 0, stream>>>(img_fake, target,
                                                  p1, t1, p2, t2, acc);
  final_kernel<<<1, 1, 0, stream>>>(acc, (float*)d_out);
}
